// Attention_20547123544149
// MI455X (gfx1250) — matrix operand order repaired and checked by reading, not compiled, not run on hardware
//
#include <hip/hip_runtime.h>
#include <hip/hip_bf16.h>
#include <math.h>

typedef __attribute__((ext_vector_type(16))) __bf16 v16bf;
typedef __attribute__((ext_vector_type(8)))  __bf16 v8bf;
typedef __attribute__((ext_vector_type(4)))  __bf16 v4bf;
typedef __attribute__((ext_vector_type(8)))  float  v8f;
typedef __attribute__((ext_vector_type(4)))  float  v4f;

#define HE   1024
#define HD   1024
#define SEQ  4096
#define NBAT 32

// Hardware tanh (V_TANH_F32 trans op) when available, ocml fallback otherwise.
#if defined(__has_builtin)
#if __has_builtin(__builtin_amdgcn_tanhf)
#define FAST_TANH(x) __builtin_amdgcn_tanhf(x)
#endif
#endif
#ifndef FAST_TANH
#define FAST_TANH(x) tanhf(x)
#endif

// ---------------- kernel 0: hb[b][j] = sum_k h[b][k]*W_h[k][j] + b_attn[j] ----
__global__ __launch_bounds__(256) void hb_kernel(const float* __restrict__ hidden,
                                                 const float* __restrict__ W,
                                                 const float* __restrict__ bias,
                                                 float* __restrict__ hb) {
    int idx = blockIdx.x * 256 + threadIdx.x;     // 32*1024 outputs
    int b = idx >> 10, j = idx & 1023;
    const float* h = hidden + (size_t)NBAT * HE + (size_t)b * HE;  // last layer
    float acc = bias[j];
#pragma unroll 4
    for (int k = 0; k < HE; ++k)
        acc = fmaf(h[k], W[(size_t)k * HD + j], acc);
    hb[idx] = acc;
}

// ---------------- kernel 1: Wt[n][k] = bf16(W_e[k][n]) (transpose+convert) ----
__global__ __launch_bounds__(256) void wt_kernel(const float* __restrict__ W,
                                                 __bf16* __restrict__ Wt) {
    int idx = blockIdx.x * 256 + threadIdx.x;     // 1M elements
    int n = idx >> 10, k = idx & 1023;
    Wt[idx] = (__bf16)W[(size_t)(HE + k) * HD + n];
}

// ---------------- kernel 2: fused e_proj GEMM + tanh + v_w dot -> scores -----
__global__ __launch_bounds__(256) void energy_kernel(const float* __restrict__ enc,
                                                     const __bf16* __restrict__ Wt,
                                                     const float* __restrict__ hb,
                                                     const float* __restrict__ vw,
                                                     float* __restrict__ scores) {
    __shared__ __attribute__((aligned(16))) __bf16 sA[32 * 40]; // 32 rows, pad to 80B
    __shared__ float sScore[32];

    const int tid  = threadIdx.x;
    const int wave = tid >> 5;
    const int lane = tid & 31;
    const int lh   = lane & 15;
    const int hi   = lane >> 4;
    const int mbase   = blockIdx.x * 32;          // 32 rows of flattened (B*S)
    const int b       = mbase >> 12;              // S = 4096 rows per batch
    const int colbase = wave * 128;               // this wave's 128 N-columns

    const float* encBase = enc + (size_t)mbase * HE;
    // Per-lane base into transposed bf16 weights: row n = colbase + lh,
// B-layout: lanes 0-15 hold K {0..7,16..23}, lanes 16-31 hold K {8..15,24..31} (interleave).
    const __bf16* wbase = Wt + (size_t)(colbase + lh) * HE + hi * 8;

    const int arow = tid >> 3;                    // cooperative A staging
    const int acol = (tid & 7) * 4;

    v8f acc[2][8];
    const v8f vzero = {0.f, 0.f, 0.f, 0.f, 0.f, 0.f, 0.f, 0.f};
#pragma unroll
    for (int mt = 0; mt < 2; ++mt)
#pragma unroll
        for (int nt = 0; nt < 8; ++nt) acc[mt][nt] = vzero;

    for (int kc = 0; kc < HE; kc += 32) {
        // stage 32x32 f32 encoder tile -> bf16 LDS; non-temporal: encoder is
        // streamed exactly once, keep Wt/hb/vw hot in cache instead.
        v4f ev = __builtin_nontemporal_load(
            (const v4f*)(encBase + (size_t)arow * HE + kc + acol));
        v4bf eb = { (__bf16)ev.x, (__bf16)ev.y, (__bf16)ev.z, (__bf16)ev.w };
        *(v4bf*)(&sA[arow * 40 + acol]) = eb;
        __syncthreads();

        // A fragments (interleave: lane<16 -> K{0..7,16..23}; lane>=16 -> K{8..15,24..31})
        v16bf afrag[2];
#pragma unroll
        for (int mt = 0; mt < 2; ++mt) {
            const __bf16* ab = &sA[(mt * 16 + lh) * 40 + hi * 8];
            ((v8bf*)&afrag[mt])[0] = *(const v8bf*)(ab);       // ds_load_b128
            ((v8bf*)&afrag[mt])[1] = *(const v8bf*)(ab + 16);  // ds_load_b128
        }

#pragma unroll
        for (int nt = 0; nt < 8; ++nt) {
            v16bf bfrag;
            const __bf16* bb = wbase + (size_t)nt * 16 * HE + kc;
            ((v8bf*)&bfrag)[0] = *(const v8bf*)(bb);           // global_load_b128
            ((v8bf*)&bfrag)[1] = *(const v8bf*)(bb + 16);
            acc[0][nt] = __builtin_amdgcn_wmma_f32_16x16x32_bf16(
                false, afrag[0], false, bfrag, (short)0, acc[0][nt], false, false);
            acc[1][nt] = __builtin_amdgcn_wmma_f32_16x16x32_bf16(
                false, afrag[1], false, bfrag, (short)0, acc[1][nt], false, false);
        }
        __syncthreads();
    }

    if (tid < 32) sScore[tid] = 0.f;
    __syncthreads();

    // per-lane column constants
    float hbv[8], vv[8];
#pragma unroll
    for (int nt = 0; nt < 8; ++nt) {
        int n = colbase + nt * 16 + lh;
        hbv[nt] = hb[b * HD + n];
        vv[nt]  = vw[n];
    }

    // D layout: lane -> N = lh, vgpr r -> M = r + hi*8
#pragma unroll
    for (int mt = 0; mt < 2; ++mt) {
#pragma unroll
        for (int r = 0; r < 8; ++r) {
            float s = 0.f;
#pragma unroll
            for (int nt = 0; nt < 8; ++nt)
                s = fmaf(FAST_TANH(acc[mt][nt][r] + hbv[nt]), vv[nt], s);
            s += __shfl_xor(s, 1);
            s += __shfl_xor(s, 2);
            s += __shfl_xor(s, 4);
            s += __shfl_xor(s, 8);      // reduce within 16-lane half
            if (lh == 0) atomicAdd(&sScore[mt * 16 + r + hi * 8], s);
        }
    }
    __syncthreads();
    if (tid < 32) scores[mbase + tid] = sScore[tid];
}

// ---------------- kernel 3: in-place softmax over S per batch row ------------
__global__ __launch_bounds__(256) void softmax_kernel(float* __restrict__ io) {
    __shared__ float red[8];
    const int tid = threadIdx.x;
    float* p = io + (size_t)blockIdx.x * SEQ;

    float x[16];
    float mx = -INFINITY;
#pragma unroll
    for (int i = 0; i < 16; ++i) { x[i] = p[tid + i * 256]; mx = fmaxf(mx, x[i]); }
#pragma unroll
    for (int o = 1; o < 32; o <<= 1) mx = fmaxf(mx, __shfl_xor(mx, o));
    if ((tid & 31) == 0) red[tid >> 5] = mx;
    __syncthreads();
    float bmax = red[0];
#pragma unroll
    for (int w = 1; w < 8; ++w) bmax = fmaxf(bmax, red[w]);
    __syncthreads();

    float sum = 0.f;
#pragma unroll
    for (int i = 0; i < 16; ++i) { x[i] = __expf(x[i] - bmax); sum += x[i]; }
#pragma unroll
    for (int o = 1; o < 32; o <<= 1) sum += __shfl_xor(sum, o);
    if ((tid & 31) == 0) red[tid >> 5] = sum;
    __syncthreads();
    float bsum = 0.f;
#pragma unroll
    for (int w = 0; w < 8; ++w) bsum += red[w];
    float inv = 1.f / bsum;
#pragma unroll
    for (int i = 0; i < 16; ++i) p[tid + i * 256] = x[i] * inv;
}

extern "C" void kernel_launch(void* const* d_in, const int* in_sizes, int n_in,
                              void* d_out, int out_size, void* d_ws, size_t ws_size,
                              hipStream_t stream) {
    const float* hidden = (const float*)d_in[0];   // (2,32,1024)
    const float* enc    = (const float*)d_in[1];   // (32,4096,1024)
    const float* W      = (const float*)d_in[2];   // (2048,1024)
    const float* bias   = (const float*)d_in[3];   // (1024,)
    const float* vw     = (const float*)d_in[4];   // (1024,)
    float* out = (float*)d_out;                    // (32,4096)

    float*  hb = (float*)d_ws;                             // 32*1024 f32 = 128 KB
    __bf16* Wt = (__bf16*)((char*)d_ws + 32 * 1024 * 4);   // 1024*1024 bf16 = 2 MB

    hb_kernel<<<dim3(128), dim3(256), 0, stream>>>(hidden, W, bias, hb);
    wt_kernel<<<dim3(4096), dim3(256), 0, stream>>>(W, Wt);
    energy_kernel<<<dim3((NBAT * SEQ) / 32), dim3(256), 0, stream>>>(enc, Wt, hb, vw, out);
    softmax_kernel<<<dim3(NBAT), dim3(256), 0, stream>>>(out);
}
